// LSTMEncoder_26946624815942
// MI455X (gfx1250) — matrix operand order repaired and checked by reading, not compiled, not run on hardware
//
#include <hip/hip_runtime.h>
#include <stdint.h>

#define B_ 32
#define T_ 2048
#define I_ 128
#define H_ 256
#define G4_ (4*H_)          // 1024 gate rows
#define NWG 16              // persistent workgroups (one N-slice each)
#define UNITS (H_/NWG)      // 16 hidden units per WG
#define THREADS 128         // 4 waves: wave w computes gate w for this slice

typedef __attribute__((ext_vector_type(16))) __bf16 v16bf;
typedef __attribute__((ext_vector_type(8)))  float  v8f;

union Frag { v16bf h; unsigned u[8]; uint4 q[2]; };

__device__ inline unsigned lane_id() { return threadIdx.x & 31u; }

__device__ inline unsigned short f2bf(float x) {
  unsigned u = __builtin_bit_cast(unsigned, x);
  u += 0x7FFFu + ((u >> 16) & 1u);          // round-to-nearest-even
  return (unsigned short)(u >> 16);
}

__device__ inline float sigmoid_f(float x) { return 1.0f / (1.0f + __expf(-x)); }
__device__ inline float tanh_f(float x) {
  float e = __expf(-2.0f * fabsf(x));
  float t = (1.0f - e) / (1.0f + e);
  return copysignf(t, x);
}

// A fragment (16x32 bf16, MxK). ISA layout: lanes 0-15 hold M=lane (kb=0),
// lanes 16-31 kb=8. VGPRs 0-3 = K pairs {kb..kb+7}, VGPRs 4-7 = {16+kb..16+kb+7}.
// Each half = 4 consecutive dwords, 16B aligned for our strides -> 2x b128.
__device__ inline Frag load_a(const unsigned short* src, int mbase, int kc, int stride) {
  Frag f; unsigned l = lane_id();
  int m  = mbase + (int)(l & 15u);
  int kb = (l < 16u) ? 0 : 8;
  int e  = m * stride + kc * 32 + kb;          // bf16-element index, multiple of 8
  const uint4* s = (const uint4*)src;
  f.q[0] = s[e >> 3];
  f.q[1] = s[(e + 16) >> 3];
  return f;
}

// B fragment (32x16 bf16, KxN): B[k][n] = W[nrow_base+n][k], W row-major, stride K.
// Lanes 0-15: n=lane, K=kc*32+{0..7,16..23}; lanes 16-31: K=kc*32+{8..15,24..31}.
__device__ inline Frag load_b(const unsigned short* w, int nrow_base, int kc, int K) {
  Frag f; unsigned l = lane_id();
  int row = nrow_base + (int)(l & 15u);
  int kb  = (l < 16u) ? 0 : 8;
  int e   = row * K + kc * 32 + kb;            // multiple of 8
  const uint4* s = (const uint4*)w;
  f.q[0] = s[e >> 3];
  f.q[1] = s[(e + 16) >> 3];
  return f;
}

// Monotonic-target grid barrier across NWG persistent workgroups.
__device__ inline void grid_barrier(unsigned* bar, unsigned target) {
  __threadfence();
  __syncthreads();
  if (threadIdx.x == 0) {
    __hip_atomic_fetch_add(bar, 1u, __ATOMIC_RELEASE, __HIP_MEMORY_SCOPE_AGENT);
    while (__hip_atomic_load(bar, __ATOMIC_ACQUIRE, __HIP_MEMORY_SCOPE_AGENT) < target)
      __builtin_amdgcn_s_sleep(1);
  }
  __syncthreads();
}

// Persistent LSTM layer scan. Recurrent + input weights stay resident in VGPRs
// (bf16 B-fragments) for all T steps; hidden state exchanged via L2-resident
// double-buffered global bf16 buffer; cell state in per-thread registers.
template<int KIH, bool LAYER0>
__global__ void __launch_bounds__(THREADS, 1)
lstm_scan_kernel(const unsigned short* __restrict__ in_bf16, long t_stride, int row_stride,
                 const unsigned short* __restrict__ w_ih,
                 const unsigned short* __restrict__ w_hh,
                 const float* __restrict__ bias,
                 unsigned short* __restrict__ h_buf,      // [2][B][H] bf16
                 unsigned short* __restrict__ out_bf16,   // layer0: [T][B][H]
                 float* __restrict__ out_f32,             // layer1: [B][T][H]
                 float* __restrict__ hN, float* __restrict__ cN,
                 unsigned* __restrict__ bar)
{
  const int wg = blockIdx.x;                 // hidden-unit slice [wg*16, wg*16+16)
  const int wv = (int)(threadIdx.x >> 5);    // gate index: 0=i 1=f 2=g 3=o
  const int u0 = wg * UNITS;
  const int growbase = wv * H_ + u0;         // global gate-row base of this wave's N-tile
  const unsigned l = lane_id();

  // ---- persistent weight fragments (registers, live across all 2048 steps) ----
  Frag Bih[KIH], Bhh[8];
  #pragma unroll
  for (int kc = 0; kc < KIH; ++kc) Bih[kc] = load_b(w_ih, growbase, kc, KIH * 32);
  #pragma unroll
  for (int kc = 0; kc < 8;   ++kc) Bhh[kc] = load_b(w_hh, growbase, kc, H_);

  const float bias_n = bias[growbase + (int)(l & 15u)];

  // zero both h double-buffers for this WG's slice
  for (int i = threadIdx.x; i < 2 * B_ * UNITS; i += THREADS) {
    int buf = i / (B_ * UNITS);
    int r   = i - buf * (B_ * UNITS);
    int b   = r / UNITS, u = r - (r / UNITS) * UNITS;
    h_buf[buf * (B_ * H_) + b * H_ + u0 + u] = 0;
  }

  __shared__ float gate_lds[4 * B_ * UNITS];   // [gate][batch][unit] = 8 KB
  float creg[4] = {0.f, 0.f, 0.f, 0.f};        // cell state, fixed (b,u) per slot

  unsigned tgt = NWG;
  grid_barrier(bar, tgt);                      // zeroed h visible everywhere

  for (int t = 0; t < T_; ++t) {
    const unsigned short* xin   = in_bf16 + (size_t)t * t_stride;
    const unsigned short* hprev = h_buf + ((t + 1) & 1) * (B_ * H_);

    v8f acc0, acc1;
    #pragma unroll
    for (int i = 0; i < 8; ++i) { acc0[i] = bias_n; acc1[i] = bias_n; }

    // input contribution: x_t[32,Kin] x W_ih^T (W_ih fragments resident)
    #pragma unroll
    for (int kc = 0; kc < KIH; ++kc) {
      Frag a0 = load_a(xin, 0,  kc, row_stride);
      Frag a1 = load_a(xin, 16, kc, row_stride);
      acc0 = __builtin_amdgcn_wmma_f32_16x16x32_bf16(false, a0.h, false, Bih[kc].h, (short)0, acc0, false, false);
      acc1 = __builtin_amdgcn_wmma_f32_16x16x32_bf16(false, a1.h, false, Bih[kc].h, (short)0, acc1, false, false);
    }
    // recurrent contribution: h[32,256] x W_hh^T (W_hh fragments resident)
    #pragma unroll
    for (int kc = 0; kc < 8; ++kc) {
      Frag a0 = load_a(hprev, 0,  kc, H_);
      Frag a1 = load_a(hprev, 16, kc, H_);
      acc0 = __builtin_amdgcn_wmma_f32_16x16x32_bf16(false, a0.h, false, Bhh[kc].h, (short)0, acc0, false, false);
      acc1 = __builtin_amdgcn_wmma_f32_16x16x32_bf16(false, a1.h, false, Bhh[kc].h, (short)0, acc1, false, false);
    }

    // scatter accumulators to LDS: C/D layout lane n=l&15, VGPR v -> M = v + 8*(l>=16)
    {
      int n  = (int)(l & 15u);
      int mo = (l >= 16u) ? 8 : 0;
      #pragma unroll
      for (int v = 0; v < 8; ++v) {
        gate_lds[(wv * B_ + (v + mo))      * UNITS + n] = acc0[v];
        gate_lds[(wv * B_ + (16 + v + mo)) * UNITS + n] = acc1[v];
      }
    }
    __syncthreads();

    if (t + 1 < T_)
      __builtin_prefetch(in_bf16 + (size_t)(t + 1) * t_stride + (threadIdx.x << 4), 0, 1);

    // elementwise LSTM cell: 128 threads x 4 = 512 = B*UNITS elements
    unsigned short* hout = h_buf + (t & 1) * (B_ * H_);
    #pragma unroll
    for (int j = 0; j < 4; ++j) {
      int idx = (int)threadIdx.x * 4 + j;
      int b = idx >> 4, u = idx & 15;
      float ig = sigmoid_f(gate_lds[(0 * B_ + b) * UNITS + u]);
      float fg = sigmoid_f(gate_lds[(1 * B_ + b) * UNITS + u]);
      float gg = tanh_f   (gate_lds[(2 * B_ + b) * UNITS + u]);
      float og = sigmoid_f(gate_lds[(3 * B_ + b) * UNITS + u]);
      float c = fg * creg[j] + ig * gg;
      creg[j] = c;
      float h = og * tanh_f(c);
      unsigned short hb = f2bf(h);
      hout[b * H_ + u0 + u] = hb;
      if (LAYER0) {
        out_bf16[((size_t)t * B_ + b) * H_ + u0 + u] = hb;        // feeds layer 1
      } else {
        out_f32[((size_t)b * T_ + t) * H_ + u0 + u] = h;          // final output
      }
      if (t == T_ - 1) {
        hN[b * H_ + u0 + u] = h;
        cN[b * H_ + u0 + u] = c;
      }
    }
    tgt += NWG;
    grid_barrier(bar, tgt);    // publishes h(t); also protects gate_lds reuse
  }
}

__global__ void zero_bar_kernel(unsigned* bar) {
  if (threadIdx.x < 4) bar[threadIdx.x] = 0u;
}

__global__ void cvt_bf16_kernel(const float* __restrict__ in, unsigned short* __restrict__ out, long n) {
  long i = (long)blockIdx.x * blockDim.x + threadIdx.x;
  if (i < n) out[i] = f2bf(in[i]);
}

__global__ void bias_sum_kernel(const float* __restrict__ a, const float* __restrict__ b,
                                float* __restrict__ o, int n) {
  int i = (int)(blockIdx.x * blockDim.x + threadIdx.x);
  if (i < n) o[i] = a[i] + b[i];
}

extern "C" void kernel_launch(void* const* d_in, const int* in_sizes, int n_in,
                              void* d_out, int out_size, void* d_ws, size_t ws_size,
                              hipStream_t stream) {
  (void)in_sizes; (void)n_in; (void)out_size; (void)ws_size;
  const float* x    = (const float*)d_in[0];
  const float* Wih0 = (const float*)d_in[1];
  const float* Whh0 = (const float*)d_in[2];
  const float* bih0 = (const float*)d_in[3];
  const float* bhh0 = (const float*)d_in[4];
  const float* Wih1 = (const float*)d_in[5];
  const float* Whh1 = (const float*)d_in[6];
  const float* bih1 = (const float*)d_in[7];
  const float* bhh1 = (const float*)d_in[8];

  char* ws = (char*)d_ws;
  size_t off = 0;
  auto wsalloc = [&](size_t bytes) -> char* {
    char* p = ws + off;
    off = (off + bytes + 255) & ~(size_t)255;
    return p;
  };

  unsigned*       bar   = (unsigned*)      wsalloc(256);
  unsigned short* xb    = (unsigned short*)wsalloc((size_t)B_ * T_ * I_ * 2);   // 16 MB
  unsigned short* wih0b = (unsigned short*)wsalloc((size_t)G4_ * I_ * 2);
  unsigned short* whh0b = (unsigned short*)wsalloc((size_t)G4_ * H_ * 2);
  unsigned short* wih1b = (unsigned short*)wsalloc((size_t)G4_ * H_ * 2);
  unsigned short* whh1b = (unsigned short*)wsalloc((size_t)G4_ * H_ * 2);
  float*          bias0 = (float*)         wsalloc(G4_ * 4);
  float*          bias1 = (float*)         wsalloc(G4_ * 4);
  unsigned short* hbuf  = (unsigned short*)wsalloc((size_t)2 * B_ * H_ * 2);
  unsigned short* out0b = (unsigned short*)wsalloc((size_t)T_ * B_ * H_ * 2);   // 32 MB

  float* out1 = (float*)d_out;                       // [B,T,H]
  float* hN   = out1 + (size_t)B_ * T_ * H_;         // [2,B,H]
  float* cN   = hN + 2 * (size_t)B_ * H_;            // [2,B,H]

  zero_bar_kernel<<<1, 32, 0, stream>>>(bar);
  long nx = (long)B_ * T_ * I_;
  cvt_bf16_kernel<<<(unsigned)((nx + 255) / 256), 256, 0, stream>>>(x, xb, nx);
  cvt_bf16_kernel<<<(G4_ * I_ + 255) / 256, 256, 0, stream>>>(Wih0, wih0b, (long)G4_ * I_);
  cvt_bf16_kernel<<<(G4_ * H_ + 255) / 256, 256, 0, stream>>>(Whh0, whh0b, (long)G4_ * H_);
  cvt_bf16_kernel<<<(G4_ * H_ + 255) / 256, 256, 0, stream>>>(Wih1, wih1b, (long)G4_ * H_);
  cvt_bf16_kernel<<<(G4_ * H_ + 255) / 256, 256, 0, stream>>>(Whh1, whh1b, (long)G4_ * H_);
  bias_sum_kernel<<<(G4_ + 255) / 256, 256, 0, stream>>>(bih0, bhh0, bias0, G4_);
  bias_sum_kernel<<<(G4_ + 255) / 256, 256, 0, stream>>>(bih1, bhh1, bias1, G4_);

  // Layer 0: input x [B,T,I] -> per-t base = t*I, batch row stride T*I
  lstm_scan_kernel<I_/32, true><<<NWG, THREADS, 0, stream>>>(
      xb, (long)I_, T_ * I_, wih0b, whh0b, bias0, hbuf,
      out0b, nullptr, hN + 0, cN + 0, bar + 0);

  // Layer 1: input out0 [T,B,H] -> per-t base = t*B*H, batch row stride H
  lstm_scan_kernel<H_/32, false><<<NWG, THREADS, 0, stream>>>(
      out0b, (long)B_ * H_, H_, wih1b, whh1b, bias1, hbuf,
      nullptr, out1, hN + (size_t)B_ * H_, cN + (size_t)B_ * H_, bar + 1);
}
